// MultiLoraConv2d_34007551050176
// MI455X (gfx1250) — matrix operand order repaired and checked by reading, not compiled, not run on hardware
//
#include <hip/hip_runtime.h>
#include <hip/hip_bf16.h>

// MultiLoRA grouped conv 3x3 as implicit-GEMM on CDNA5 WMMA (bf16 -> f32 acc),
// with async global->LDS staging (ASYNCcnt) for the input halo tile.
// B=8, C_IN=C_OUT=320, H=W=64, K(reduction)=320*9=2880.

#define BATCH   8
#define CH      320
#define HW      64
#define PIX     (HW*HW)          // 4096
#define CP      (CH/2)           // 160 channel-pairs
#define CHUNKS  10               // 320/32 ci chunks
#define MSUBS   20               // 320/16 M subtiles
#define MTILES  5                // 320/64 per-workgroup M tiles

typedef __attribute__((ext_vector_type(4)))  unsigned int v4u;
typedef __attribute__((ext_vector_type(8)))  unsigned int v8u;
typedef __attribute__((ext_vector_type(16))) __bf16      v16bf;
typedef __attribute__((ext_vector_type(8)))  float       v8f;

__device__ __forceinline__ unsigned int f2bf(float f) {
    unsigned int u = __builtin_bit_cast(unsigned int, f);
    u += 0x7FFFu + ((u >> 16) & 1u);      // round-to-nearest-even
    return u >> 16;
}

// ---- pack x: fp32 [B][320][64][64] -> bf16-pair dwords [B][160][64][64] ----
__global__ __launch_bounds__(256) void pack_x_kernel(const float* __restrict__ x,
                                                     unsigned int* __restrict__ xpack) {
    size_t i = (size_t)blockIdx.x * 256 + threadIdx.x;   // B*CP*PIX = 5,242,880
    if (i >= (size_t)BATCH * CP * PIX) return;
    unsigned int pix = (unsigned int)(i % PIX);
    unsigned int cpb = (unsigned int)(i / PIX);
    unsigned int cp  = cpb % CP;
    unsigned int b   = cpb / CP;
    size_t src = ((size_t)(b * CH + cp * 2)) * PIX + pix;
    unsigned int lo = f2bf(x[src]);
    unsigned int hi = f2bf(x[src + PIX]);
    xpack[i] = lo | (hi << 16);
}

// ---- pack weights into WMMA A-fragment layout -------------------------------
// wpack dword index = ((((b*20 + msub)*9 + kpos)*10 + cc)*32 + lane)*8 + v
// lane l (g = l>>4, m = l&15): row M = msub*16 + m
//   K = 16g + 2v (+1)
__global__ __launch_bounds__(256) void pack_w_kernel(const float* __restrict__ weight,
                                                     const int* __restrict__ adapter_ids,
                                                     unsigned int* __restrict__ wpack) {
    size_t i = (size_t)blockIdx.x * 256 + threadIdx.x;   // 8*20*9*10*256 = 3,686,400
    if (i >= (size_t)BATCH * MSUBS * 9 * CHUNKS * 256) return;
    unsigned int v    = (unsigned int)(i & 7);
    unsigned int lane = (unsigned int)((i >> 3) & 31);
    unsigned int rest = (unsigned int)(i >> 8);
    unsigned int cc   = rest % CHUNKS;  rest /= CHUNKS;
    unsigned int kpos = rest % 9;       rest /= 9;
    unsigned int msub = rest % MSUBS;
    unsigned int b    = rest / MSUBS;

    int a = adapter_ids[b];
    unsigned int M  = msub * 16 + (lane & 15);
    unsigned int g  = lane >> 4;
    unsigned int k0 = g * 16 + 2 * v;
    unsigned int ci = cc * 32 + k0;
    unsigned int kh = kpos / 3, kw = kpos % 3;
    size_t base = ((((size_t)a * CH + M) * CH + ci) * 3 + kh) * 3 + kw;
    unsigned int lo = f2bf(weight[base]);       // ci
    unsigned int hi = f2bf(weight[base + 9]);   // ci+1 (ci-stride = 9 floats)
    wpack[i] = lo | (hi << 16);
}

// ---- main implicit-GEMM conv -----------------------------------------------
// grid = (16 pixel blocks [4 x * 4 y], 5 M-tiles, 8 batches), 256 threads.
// Workgroup tile: 64 out-channels x (16 rows x 16 cols) pixels.
// Wave w computes output rows {2w, 2w+1} for 64 output channels (8 C-frags).
#define LROWS 18
#define LCOLS 18
#define LCPS  20            // 16 ci-pairs padded to 20 dwords (16B aligned)
#define NPOS  (LROWS*LCOLS) // 324 halo positions
#define LDS_DW (NPOS*LCPS)

__global__ __launch_bounds__(256) void conv_wmma_kernel(const unsigned int* __restrict__ xpack,
                                                        const unsigned int* __restrict__ wpack,
                                                        float* __restrict__ out) {
    __shared__ unsigned int xs[LDS_DW];   // 25,920 bytes

    const unsigned int tid     = threadIdx.x;
    const unsigned int lane    = tid & 31;
    const unsigned int laneLo  = lane & 15;
    const unsigned int laneHi  = lane >> 4;
    const unsigned int wave    = tid >> 5;                 // 0..7
    const unsigned int owBase  = (blockIdx.x & 3) * 16;
    const unsigned int ohBase  = (blockIdx.x >> 2) * 16;
    const unsigned int mtile   = blockIdx.y;               // 0..4 (64 channels each)
    const unsigned int b       = blockIdx.z;

    v8f acc[8] = {};   // [m*2 + row] ; fully unrolled -> registers

    // ---- precompute this thread's halo staging positions (fixed across cc) --
    const int p0 = (int)tid;              // always < NPOS
    const int p1 = (int)tid + 256;        // valid if < NPOS
    const int r0p = p0 / LCOLS, c0p = p0 % LCOLS;
    const int r1p = p1 / LCOLS, c1p = p1 % LCOLS;
    const int gr0 = (int)ohBase + r0p - 1, gc0 = (int)owBase + c0p - 1;
    const int gr1 = (int)ohBase + r1p - 1, gc1 = (int)owBase + c1p - 1;
    const bool ok0 = (gr0 >= 0 && gr0 < HW && gc0 >= 0 && gc0 < HW);
    const bool ok1 = (gr1 >= 0 && gr1 < HW && gc1 >= 0 && gc1 < HW);
    const unsigned int ldsIdx0 = (unsigned int)(p0 * LCPS);
    const unsigned int ldsIdx1 = (unsigned int)(p1 * LCPS);
    const long long gofs0 = ((long long)gr0 * HW + gc0) * 4;   // bytes in plane
    const long long gofs1 = ((long long)gr1 * HW + gc1) * 4;

    const size_t xbBase = (size_t)b * CP * PIX;                         // dwords
    const size_t wbBase = (((size_t)b * MSUBS + mtile * 4) * 9) * CHUNKS * 256;

    for (unsigned int cc = 0; cc < CHUNKS; ++cc) {
        // -------- stage halo tile [18r][18c][16 ci-pairs] via async copies ---
        const unsigned long long gplane =
            (unsigned long long)(uintptr_t)xpack +
            (unsigned long long)(xbBase + (size_t)cc * 16 * PIX) * 4;

        if (ok0) {
            unsigned int l = (unsigned int)(uintptr_t)(xs + ldsIdx0);
            unsigned long long g = gplane + (unsigned long long)gofs0;
            #pragma unroll
            for (int cp = 0; cp < 16; ++cp) {
                asm volatile("global_load_async_to_lds_b32 %0, %1, off"
                             :: "v"(l), "v"(g) : "memory");
                l += 4; g += (unsigned long long)PIX * 4;
            }
        } else {
            #pragma unroll
            for (int cp = 0; cp < 16; ++cp) xs[ldsIdx0 + cp] = 0u;
        }
        if (p1 < NPOS) {
            if (ok1) {
                unsigned int l = (unsigned int)(uintptr_t)(xs + ldsIdx1);
                unsigned long long g = gplane + (unsigned long long)gofs1;
                #pragma unroll
                for (int cp = 0; cp < 16; ++cp) {
                    asm volatile("global_load_async_to_lds_b32 %0, %1, off"
                                 :: "v"(l), "v"(g) : "memory");
                    l += 4; g += (unsigned long long)PIX * 4;
                }
            } else {
                #pragma unroll
                for (int cp = 0; cp < 16; ++cp) xs[ldsIdx1 + cp] = 0u;
            }
        }
        asm volatile("s_wait_asynccnt 0" ::: "memory");
        __syncthreads();

        // -------- 9 kernel taps * 4 M-subtiles * 2 rows of WMMA --------------
        #pragma unroll
        for (unsigned int kh = 0; kh < 3; ++kh) {
            #pragma unroll
            for (unsigned int kw = 0; kw < 3; ++kw) {
                const unsigned int r0  = 2 * wave + kh;    // 0..17
                const unsigned int col = laneLo + kw;      // 0..17
                const unsigned int* bp0 = &xs[(r0 * LCOLS + col) * LCPS + 8 * laneHi];
                const unsigned int* bp1 = bp0 + LCOLS * LCPS;   // next output row
                v8u braw0, braw1;
                braw0.lo = *(const v4u*)(bp0);
                braw0.hi = *(const v4u*)(bp0 + 4);
                braw1.lo = *(const v4u*)(bp1);
                braw1.hi = *(const v4u*)(bp1 + 4);
                v16bf bf0 = __builtin_bit_cast(v16bf, braw0);
                v16bf bf1 = __builtin_bit_cast(v16bf, braw1);

                const size_t kidx = ((size_t)(kh * 3 + kw) * CHUNKS + cc) * 256 + lane * 8;
                #pragma unroll
                for (unsigned int m = 0; m < 4; ++m) {
                    const unsigned int* ap =
                        &wpack[wbBase + (size_t)m * 9 * CHUNKS * 256 + kidx];
                    v8u araw;
                    araw.lo = *(const v4u*)(ap);
                    araw.hi = *(const v4u*)(ap + 4);
                    v16bf af = __builtin_bit_cast(v16bf, araw);
                    acc[2 * m] = __builtin_amdgcn_wmma_f32_16x16x32_bf16(
                        false, af, false, bf0, (short)0, acc[2 * m], false, false);
                    acc[2 * m + 1] = __builtin_amdgcn_wmma_f32_16x16x32_bf16(
                        false, af, false, bf1, (short)0, acc[2 * m + 1], false, false);
                }
            }
        }
        __syncthreads();
    }

    // -------- epilogue: C layout -> out[b][ch][oh][ow] (fp32) ---------------
    const unsigned int oh0 = ohBase + 2 * wave;
    const unsigned int ow  = owBase + laneLo;
    #pragma unroll
    for (unsigned int m = 0; m < 4; ++m) {
        const unsigned int ch0 = mtile * 64 + m * 16 + 8 * laneHi;
        #pragma unroll
        for (unsigned int r = 0; r < 8; ++r) {
            const size_t chOfs = ((size_t)b * CH + ch0 + r) * PIX + ow;
            out[chOfs + (size_t)oh0 * HW]       = acc[2 * m][r];
            out[chOfs + (size_t)(oh0 + 1) * HW] = acc[2 * m + 1][r];
        }
    }
}

extern "C" void kernel_launch(void* const* d_in, const int* in_sizes, int n_in,
                              void* d_out, int out_size, void* d_ws, size_t ws_size,
                              hipStream_t stream) {
    const float* x           = (const float*)d_in[0];
    const float* weight      = (const float*)d_in[1];
    const int*   adapter_ids = (const int*)d_in[2];
    float*       out         = (float*)d_out;

    unsigned int* xpack = (unsigned int*)d_ws;                       // 20.97 MB
    unsigned int* wpack = xpack + (size_t)BATCH * CP * PIX;          // 14.75 MB

    const unsigned int nx = BATCH * CP * PIX;                        // 5,242,880
    const unsigned int nw = BATCH * MSUBS * 9 * CHUNKS * 256;        // 3,686,400
    pack_x_kernel<<<nx / 256, 256, 0, stream>>>(x, xpack);
    pack_w_kernel<<<nw / 256, 256, 0, stream>>>(weight, adapter_ids, wpack);

    dim3 grid(16, MTILES, BATCH);
    conv_wmma_kernel<<<grid, 256, 0, stream>>>(xpack, wpack, out);
}
